// UpSample_56770877718582
// MI455X (gfx1250) — matrix operand order repaired and checked by reading, not compiled, not run on hardware
//
#include <hip/hip_runtime.h>

typedef __attribute__((ext_vector_type(16))) _Float16 v16h;
typedef __attribute__((ext_vector_type(8)))  float    v8f;
typedef __attribute__((ext_vector_type(4)))  float    v4f;

#define C_IN   32
#define C_OUT  16
#define K3     8
#define NCOL   128            // K3 * C_OUT
#define STATS_BLOCKS 128
#define EPS    1e-5f

// ---------------------------------------------------------------------------
// Stage 1: per-block, per-channel partial sums of x and x^2.
// Deterministic: fixed ascending per-thread accumulation order + LDS tree
// reduction (no float atomics -> bit-stable across graph replays).
// Partials: ws[128 + b*64 + c] = sum, ws[128 + b*64 + 32 + c] = sumsq.
// ---------------------------------------------------------------------------
__global__ void __launch_bounds__(256) bn_stats(const float* __restrict__ x,
                                                float* __restrict__ ws, int n) {
    __shared__ float sSum[256];
    __shared__ float sSq[256];
    const int tid = threadIdx.x;
    const int c   = tid & 31;
    const int g0  = (blockIdx.x * 256 + tid) >> 5;          // row-group start
    const int gs  = (STATS_BLOCKS * 256) >> 5;              // row stride
    float s = 0.f, s2 = 0.f;
    for (int r = g0; r < n; r += gs) {                       // coalesced: wave = one row
        float v = x[(size_t)r * C_IN + c];
        s  += v;
        s2 += v * v;
    }
    sSum[tid] = s;
    sSq[tid]  = s2;
    __syncthreads();
    #pragma unroll
    for (int st = 128; st >= 32; st >>= 1) {
        if (tid < st) {
            sSum[tid] += sSum[tid + st];
            sSq[tid]  += sSq[tid + st];
        }
        __syncthreads();
    }
    if (tid < 32) {
        ws[128 + blockIdx.x * 64 + tid]      = sSum[tid];
        ws[128 + blockIdx.x * 64 + 32 + tid] = sSq[tid];
    }
}

// ---------------------------------------------------------------------------
// Stage 2: fixed-order sum of block partials, then scale/shift.
// ws[64+c] = scale = gamma*rsqrt(var+eps); ws[96+c] = shift = beta - mean*scale
// ---------------------------------------------------------------------------
__global__ void bn_finalize(const float* __restrict__ gamma,
                            const float* __restrict__ beta,
                            float* __restrict__ ws, float invN) {
    __shared__ float tot[64];
    const int i = threadIdx.x;            // 0..63
    float s = 0.f;
    for (int b = 0; b < STATS_BLOCKS; ++b)
        s += ws[128 + b * 64 + i];
    tot[i] = s;
    __syncthreads();
    if (i < 32) {
        float mean = tot[i] * invN;
        float var  = tot[32 + i] * invN - mean * mean;
        float sc   = gamma[i] * rsqrtf(var + EPS);
        ws[64 + i] = sc;
        ws[96 + i] = beta[i] - mean * sc;
    }
}

// ---------------------------------------------------------------------------
// Fused BN + ReLU + GEMM:  out[16M,128] tile-wise via v_wmma_f32_16x16x32_f16.
// One wave per 16-row tile; the whole K=32 reduction is ONE WMMA per column
// tile j (j == kernel-offset k3, columns 16j..16j+15 == d).
// B fragments (f16) and BN params are register-resident across the tile loop.
// ---------------------------------------------------------------------------
__global__ void __launch_bounds__(256) fused_gemm(const float* __restrict__ x,
                                                  const float* __restrict__ W,
                                                  const float* __restrict__ ws,
                                                  float* __restrict__ out,
                                                  int ntiles) {
    const int lane = threadIdx.x & 31;
    const int n    = lane & 15;      // A: row-in-tile ; B/D: column-in-tile
    const int h    = lane >> 4;      // half-wave select

    // ---- Preload B fragments: B[j][e] = W[j][c = 8h + e (+8 for e>=8)][d = n] as f16.
    // 16-bit B 32x16 layout: lane<16 -> K=0..7 then 16..23 ; lane>=16 -> K=8..15 then 24..31.
    v16h B[K3];
    #pragma unroll
    for (int j = 0; j < K3; ++j) {
        const float* wp = W + j * (C_IN * C_OUT) + (8 * h) * C_OUT + n;
        #pragma unroll
        for (int e = 0; e < 16; ++e)
            B[j][e] = (_Float16)wp[(e + (e >= 8 ? 8 : 0)) * C_OUT];
    }

    // ---- BN params for the 16 channels this lane's A fragment covers.
    // 16-bit A 16x32 layout: lane<16 -> K=0..7 then 16..23 ; lane>=16 -> K=8..15 then 24..31.
    float sc[16], sh[16];
    #pragma unroll
    for (int i = 0; i < 8; ++i) {
        sc[i]     = ws[64 + 8 * h + i];
        sh[i]     = ws[96 + 8 * h + i];
        sc[8 + i] = ws[64 + 16 + 8 * h + i];
        sh[8 + i] = ws[96 + 16 + 8 * h + i];
    }

    const int wave   = (blockIdx.x * blockDim.x + threadIdx.x) >> 5;
    const int nwaves = (gridDim.x * blockDim.x) >> 5;

    for (int t = wave; t < ntiles; t += nwaves) {
        const size_t row = (size_t)t * 16 + n;               // A-matrix M = lane&15
        const float* xr  = x + row * C_IN + 8 * h;           // channels 8h..8h+7, +16

        if (t + nwaves < ntiles)                              // global_prefetch_b8
            __builtin_prefetch(xr + (size_t)nwaves * 16 * C_IN, 0, 0);

        v4f c0 = *(const v4f*)(xr);
        v4f c1 = *(const v4f*)(xr + 4);
        v4f c2 = *(const v4f*)(xr + 16);
        v4f c3 = *(const v4f*)(xr + 20);

        v16h a;
        #pragma unroll
        for (int i = 0; i < 4; ++i) {
            a[i]      = (_Float16)fmaxf(fmaf(c0[i], sc[i],      sh[i]),      0.f);
            a[4 + i]  = (_Float16)fmaxf(fmaf(c1[i], sc[4 + i],  sh[4 + i]),  0.f);
            a[8 + i]  = (_Float16)fmaxf(fmaf(c2[i], sc[8 + i],  sh[8 + i]),  0.f);
            a[12 + i] = (_Float16)fmaxf(fmaf(c3[i], sc[12 + i], sh[12 + i]), 0.f);
        }

        float* o = out + (size_t)t * (16 * NCOL);
        #pragma unroll
        for (int j = 0; j < K3; ++j) {
            v8f acc = {};
            acc = __builtin_amdgcn_wmma_f32_16x16x32_f16(
                false, a, false, B[j], (short)0, acc, false, false);
            // C/D layout: VGPR r -> row r (lanes 0-15) / r+8 (lanes 16-31), col = lane&15.
            // Each store = two contiguous 64B segments (rows r and r+8, cols 16j..16j+15).
            #pragma unroll
            for (int r = 0; r < 8; ++r)
                o[(size_t)(r + 8 * h) * NCOL + j * C_OUT + n] = acc[r];
        }
    }
}

// ---------------------------------------------------------------------------
// Scalar tail for N % 16 rows (N=2M is divisible by 16, so normally unused).
// ---------------------------------------------------------------------------
__global__ void tail_kernel(const float* __restrict__ x, const float* __restrict__ W,
                            const float* __restrict__ ws, float* __restrict__ out,
                            int rowStart, int N) {
    int idx  = blockIdx.x * blockDim.x + threadIdx.x;
    int nrem = N - rowStart;
    if (idx >= nrem * NCOL) return;
    int r  = rowStart + idx / NCOL;
    int kd = idx % NCOL;
    int k  = kd >> 4, d = kd & 15;
    float acc = 0.f;
    for (int c = 0; c < C_IN; ++c) {
        float y = fmaxf(fmaf(x[(size_t)r * C_IN + c], ws[64 + c], ws[96 + c]), 0.f);
        acc = fmaf(y, W[k * (C_IN * C_OUT) + c * C_OUT + d], acc);
    }
    out[(size_t)r * NCOL + kd] = acc;
}

extern "C" void kernel_launch(void* const* d_in, const int* in_sizes, int n_in,
                              void* d_out, int out_size, void* d_ws, size_t ws_size,
                              hipStream_t stream) {
    const float* x     = (const float*)d_in[0];
    const float* gamma = (const float*)d_in[1];
    const float* beta  = (const float*)d_in[2];
    const float* W     = (const float*)d_in[3];
    float*       out   = (float*)d_out;
    float*       ws    = (float*)d_ws;

    const int N = in_sizes[0] / C_IN;           // 2,000,000

    bn_stats<<<STATS_BLOCKS, 256, 0, stream>>>(x, ws, N);
    bn_finalize<<<1, 64, 0, stream>>>(gamma, beta, ws, 1.0f / (float)N);

    const int ntiles = N / 16;
    fused_gemm<<<2048, 256, 0, stream>>>(x, W, ws, out, ntiles);

    const int rem = N & 15;
    if (rem) {
        int work = rem * NCOL;
        tail_kernel<<<(work + 255) / 256, 256, 0, stream>>>(x, W, ws, out, ntiles * 16, N);
    }
}
